// WordAttention_22780506538215
// MI455X (gfx1250) — matrix operand order repaired and checked by reading, not compiled, not run on hardware
//
#include <hip/hip_runtime.h>
#include <hip/hip_bf16.h>

// ---------------------------------------------------------------------------
// Fused QKV + flash attention for MI455X (gfx1250, wave32, WMMA f16->f32).
//   x:[8,2048,512] f32, W*:[512,512] f32, b*:[512] f32 -> out:[8,2048,512] f32
// ---------------------------------------------------------------------------

typedef _Float16 h8   __attribute__((ext_vector_type(8)));
typedef _Float16 v16h __attribute__((ext_vector_type(16)));
typedef float    v8f  __attribute__((ext_vector_type(8)));
typedef float    f4   __attribute__((ext_vector_type(4)));

#define LOG2E 1.44269504088896340736f

static __device__ __forceinline__ v16h make_frag(const _Float16* p0, const _Float16* p1) {
  h8 lo = *(const h8*)p0;       // 16B -> global_load_b128 / ds_load_b128
  h8 hi = *(const h8*)p1;
  return __builtin_shufflevector(lo, hi, 0,1,2,3,4,5,6,7,8,9,10,11,12,13,14,15);
}

static __device__ __forceinline__ v8f wmma_f16(v16h a, v16h b, v8f c) {
  return __builtin_amdgcn_wmma_f32_16x16x32_f16(false, a, false, b, (short)0, c, false, false);
}

// --------------------------- pack kernels ----------------------------------

__global__ void pack_x_kernel(const float* __restrict__ x, _Float16* __restrict__ xh, int n) {
  int i = blockIdx.x * blockDim.x + threadIdx.x;
  if (i < n) xh[i] = (_Float16)x[i];
}

// Wt[z][h][f] = W_z[f][h]  (n-major so GEMM B-frag reads are contiguous in K)
__global__ void pack_w_kernel(const float* __restrict__ Wq, const float* __restrict__ Wk,
                              const float* __restrict__ Wv, _Float16* __restrict__ Wt) {
  int z = blockIdx.y;
  const float* W = (z == 0) ? Wq : (z == 1) ? Wk : Wv;
  int i = blockIdx.x * blockDim.x + threadIdx.x;   // 0 .. 512*512-1
  int h = i >> 9, f = i & 511;
  Wt[(size_t)z * 512 * 512 + i] = (_Float16)W[f * 512 + h];
}

// --------------------------- QKV projection --------------------------------
// Grid: (16384/64, 512/128, 3). Block 128 (4 waves). 64x128 tile, K=512.
// Each A fragment feeds 8 WMMAs (8 column tiles) -> high intensity.
// z=0 -> Q row-major, z=1 -> K row-major, z=2 -> V transposed [b][h][s].

__global__ __launch_bounds__(128) void qkv_kernel(
    const _Float16* __restrict__ xh, const _Float16* __restrict__ Wt,
    const float* __restrict__ bq, const float* __restrict__ bk, const float* __restrict__ bv,
    _Float16* __restrict__ Qh, _Float16* __restrict__ Kh, _Float16* __restrict__ Vt)
{
  const int z    = blockIdx.z;
  const int m0   = blockIdx.x * 64;
  const int n0   = blockIdx.y * 128;
  const int wv   = threadIdx.x >> 5;
  const int lane = threadIdx.x & 31;
  const int lm   = lane & 15;
  const int hi   = lane >> 4;

  const _Float16* W   = Wt + (size_t)z * 512 * 512;
  const float*   bias = (z == 0) ? bq : (z == 1) ? bk : bv;

  const int mrow = m0 + wv * 16 + lm;
  v8f acc[8] = {};

  for (int k0 = 0; k0 < 512; k0 += 32) {
    // A 16x32 f16: lanes<16 hold K 0..7 & 16..23, lanes>=16 hold K 8..15 & 24..31
    const _Float16* pa = xh + (size_t)mrow * 512 + k0 + 8 * hi;
    v16h a = make_frag(pa, pa + 16);
#pragma unroll
    for (int c = 0; c < 8; ++c) {
      // B 32x16 f16: lane = column n, lanes<16 K 0..7 & 16..23, lanes>=16 K 8..15 & 24..31
      const _Float16* pb = W + (size_t)(n0 + c * 16 + lm) * 512 + k0 + 8 * hi;
      v16h b = make_frag(pb, pb + 16);
      acc[c] = wmma_f16(a, b, acc[c]);
    }
  }

  if (z < 2) {
    _Float16* Out = (z == 0) ? Qh : Kh;
#pragma unroll
    for (int c = 0; c < 8; ++c) {
      int n = n0 + c * 16 + lm;
      float bb = bias[n];
#pragma unroll
      for (int r = 0; r < 8; ++r) {
        int m = m0 + wv * 16 + r + 8 * hi;              // C layout: M = r (+8 hi)
        Out[(size_t)m * 512 + n] = (_Float16)(acc[c][r] + bb);
      }
    }
  } else {
    // transposed store: per lane N fixed, M contiguous -> one b128 per tile
    int mg = m0 + wv * 16;
    int bidx = mg >> 11;
    int s0 = (mg & 2047) + 8 * hi;
#pragma unroll
    for (int c = 0; c < 8; ++c) {
      int n = n0 + c * 16 + lm;
      float bb = bias[n];
      h8 v;
#pragma unroll
      for (int r = 0; r < 8; ++r) v[r] = (_Float16)(acc[c][r] + bb);
      *(h8*)(Vt + ((size_t)bidx * 512 + n) * 2048 + s0) = v;
    }
  }
}

// --------------------------- flash attention -------------------------------
// Grid: (2048/32, 8). Block 128 = 4 waves cooperating on TWO 16-query tiles.
// Wave w owns h in [128w,128w+128). Each K/V fragment load feeds 2 WMMAs
// (one per q-tile): 16 frag loads -> 32 WMMAs per iteration per wave.

__global__ __launch_bounds__(128) void attn_kernel(
    const _Float16* __restrict__ Qh, const _Float16* __restrict__ Kh,
    const _Float16* __restrict__ Vt, float* __restrict__ out)
{
  __shared__ float    sS[4][2][2][16][16];  // [wave][qtile][ktile][q][k'_local]
  __shared__ _Float16 sP[4][2][16][32];     // per-wave P tiles [q][k']

  const int b    = blockIdx.y;
  const int q0   = blockIdx.x * 32;
  const int wv   = threadIdx.x >> 5;
  const int lane = threadIdx.x & 31;
  const int lm   = lane & 15;
  const int hi   = lane >> 4;
  const int hb   = wv * 128;

  const _Float16* Q = Qh + (size_t)b * 2048 * 512;
  const _Float16* K = Kh + (size_t)b * 2048 * 512;
  const _Float16* V = Vt + (size_t)b * 512 * 2048;

  // Q^T B-frags for both q-tiles over this wave's 128-wide h chunk (persistent)
  v16h qf[2][4];
#pragma unroll
  for (int u = 0; u < 2; ++u)
#pragma unroll
    for (int j = 0; j < 4; ++j) {
      const _Float16* p = Q + (size_t)(q0 + 16 * u + lm) * 512 + hb + 32 * j + 8 * hi;
      qf[u][j] = make_frag(p, p + 16);
    }

  v8f   oacc[2][8] = {};
  float m_run[2] = {-3.0e38f, -3.0e38f};
  float l_run[2] = {0.0f, 0.0f};

  for (int k0 = 0; k0 < 2048; k0 += 32) {
    // 1) partial S^T = K * Q^T over own h-chunk; K frag shared by both q-tiles
    v8f st[2][2] = {};
#pragma unroll
    for (int j = 0; j < 4; ++j) {
#pragma unroll
      for (int t = 0; t < 2; ++t) {
        const _Float16* p = K + (size_t)(k0 + 16 * t + lm) * 512 + hb + 32 * j + 8 * hi;
        v16h a = make_frag(p, p + 16);
#pragma unroll
        for (int u = 0; u < 2; ++u)
          st[u][t] = wmma_f16(a, qf[u][j], st[u][t]);
      }
    }

    // 2) cross-wave reduce of partial scores through LDS
    __syncthreads();
#pragma unroll
    for (int u = 0; u < 2; ++u)
#pragma unroll
      for (int t = 0; t < 2; ++t) {
        f4 lo = __builtin_shufflevector(st[u][t], st[u][t], 0, 1, 2, 3);
        f4 hh = __builtin_shufflevector(st[u][t], st[u][t], 4, 5, 6, 7);
        *(f4*)&sS[wv][u][t][lm][8 * hi]     = lo;
        *(f4*)&sS[wv][u][t][lm][8 * hi + 4] = hh;
      }
    __syncthreads();

    // 3) per q-tile: reduce partials, online softmax, relayout P via LDS
    v16h bp[2];
#pragma unroll
    for (int u = 0; u < 2; ++u) {
      v8f s[2];
#pragma unroll
      for (int t = 0; t < 2; ++t) {
        f4 lo = {}, hh = {};
#pragma unroll
        for (int w2 = 0; w2 < 4; ++w2) {
          lo += *(const f4*)&sS[w2][u][t][lm][8 * hi];
          hh += *(const f4*)&sS[w2][u][t][lm][8 * hi + 4];
        }
        s[t] = __builtin_shufflevector(lo, hh, 0, 1, 2, 3, 4, 5, 6, 7);
      }

      // softmax row (fixed q) lives in this lane's regs + lane^16
      float mx = s[0][0];
#pragma unroll
      for (int t = 0; t < 2; ++t)
#pragma unroll
        for (int r = 0; r < 8; ++r) mx = fmaxf(mx, s[t][r]);
      mx = fmaxf(mx, __shfl_xor(mx, 16, 32));
      float m_new = fmaxf(m_run[u], mx);
      float corr  = exp2f((m_run[u] - m_new) * LOG2E);

      float ssum = 0.0f;
      h8 p16[2];
#pragma unroll
      for (int t = 0; t < 2; ++t)
#pragma unroll
        for (int r = 0; r < 8; ++r) {
          float e = exp2f((s[t][r] - m_new) * LOG2E);
          ssum += e;
          p16[t][r] = (_Float16)e;
        }
      ssum += __shfl_xor(ssum, 16, 32);
      l_run[u] = l_run[u] * corr + ssum;
      m_run[u] = m_new;

#pragma unroll
      for (int j = 0; j < 8; ++j) oacc[u][j] *= corr;

      // C-layout P -> LDS [q][k'] (contiguous b128 both ways), read as B-frag
#pragma unroll
      for (int t = 0; t < 2; ++t)
        *(h8*)&sP[wv][u][lm][16 * t + 8 * hi] = p16[t];
      const _Float16* pb = &sP[wv][u][lm][8 * hi];
      bp[u] = make_frag(pb, pb + 16);
    }

    // 4) O^T += V^T * P over own h chunk; V frag shared by both q-tiles
#pragma unroll
    for (int j = 0; j < 8; ++j) {
      const _Float16* pv = V + (size_t)(hb + 16 * j + lm) * 2048 + k0 + 8 * hi;
      v16h av = make_frag(pv, pv + 16);
#pragma unroll
      for (int u = 0; u < 2; ++u)
        oacc[u][j] = wmma_f16(av, bp[u], oacc[u][j]);
    }
  }

  // 5) normalize and store (h contiguous per lane -> b128 stores)
#pragma unroll
  for (int u = 0; u < 2; ++u) {
    float inv_l = 1.0f / l_run[u];
#pragma unroll
    for (int j = 0; j < 8; ++j) {
      v8f o = oacc[u][j] * inv_l;
      int h0 = hb + 16 * j + 8 * hi;
      float* po = out + ((size_t)(b * 2048 + q0 + 16 * u + lm)) * 512 + h0;
      *(f4*)(po)     = __builtin_shufflevector(o, o, 0, 1, 2, 3);
      *(f4*)(po + 4) = __builtin_shufflevector(o, o, 4, 5, 6, 7);
    }
  }
}

// --------------------------- launcher --------------------------------------

extern "C" void kernel_launch(void* const* d_in, const int* in_sizes, int n_in,
                              void* d_out, int out_size, void* d_ws, size_t ws_size,
                              hipStream_t stream) {
  const float* x  = (const float*)d_in[0];
  const float* Wq = (const float*)d_in[1];
  const float* bq = (const float*)d_in[2];
  const float* Wk = (const float*)d_in[3];
  const float* bk = (const float*)d_in[4];
  const float* Wv = (const float*)d_in[5];
  const float* bv = (const float*)d_in[6];
  float* out = (float*)d_out;

  char* ws = (char*)d_ws;
  _Float16* xh = (_Float16*)(ws);                          // 16 MB: x in f16
  _Float16* Wt = (_Float16*)(ws + (16ull << 20));          // 1.5 MB: 3x [h][f]
  _Float16* Q  = (_Float16*)(ws + (18ull << 20));          // 16 MB: [b][s][h]
  _Float16* K  = (_Float16*)(ws + (35ull << 20));          // 16 MB: [b][s][h]
  _Float16* V  = (_Float16*)(ws + (52ull << 20));          // 16 MB: [b][h][s]

  const int NX = 8 * 2048 * 512;
  pack_x_kernel<<<dim3(NX / 256), 256, 0, stream>>>(x, xh, NX);
  pack_w_kernel<<<dim3(512 * 512 / 256, 3), 256, 0, stream>>>(Wq, Wk, Wv, Wt);
  qkv_kernel<<<dim3(16384 / 64, 512 / 128, 3), 128, 0, stream>>>(xh, Wt, bq, bk, bv, Q, K, V);
  attn_kernel<<<dim3(2048 / 32, 8), 128, 0, stream>>>(Q, K, V, out);
}
